// SelfAttention_87101936763184
// MI455X (gfx1250) — matrix operand order repaired and checked by reading, not compiled, not run on hardware
//
#include <hip/hip_runtime.h>
#include <hip/hip_bf16.h>

// ---------------------------------------------------------------------------
// Types for CDNA5 WMMA
// ---------------------------------------------------------------------------
typedef __attribute__((ext_vector_type(16))) _Float16 v16h;
typedef __attribute__((ext_vector_type(8)))  _Float16 v8h;
typedef __attribute__((ext_vector_type(4)))  _Float16 v4h;
typedef __attribute__((ext_vector_type(8)))  float    v8f;
typedef __attribute__((ext_vector_type(4)))  float    v4f;
typedef __attribute__((ext_vector_type(4)))  int      v4i;

#define DIM       1024
#define NHEADS    16
#define HDIM      64
#define SEQ       2048
#define BATCH     4
#define TOKENS    (BATCH * SEQ)     // 8192
#define QKV_N     (3 * DIM)         // 3072

#define AS1 __attribute__((address_space(1)))
#define AS3 __attribute__((address_space(3)))

#if defined(__has_builtin)
#if __has_builtin(__builtin_amdgcn_global_load_async_to_lds_b128) && \
    __has_builtin(__builtin_amdgcn_s_wait_asynccnt)
#define USE_ASYNC_LDS 1
#else
#define USE_ASYNC_LDS 0
#endif
#else
#define USE_ASYNC_LDS 0
#endif

__device__ __forceinline__ v8f wmma_f16(v16h a, v16h b, v8f c) {
    return __builtin_amdgcn_wmma_f32_16x16x32_f16(
        /*neg_a=*/false, a, /*neg_b=*/false, b,
        /*c_mod=*/(short)0, c, /*reuse_a=*/false, /*reuse_b=*/false);
}

__device__ __forceinline__ v8f zero8() {
    v8f z;
#pragma unroll
    for (int i = 0; i < 8; ++i) z[i] = 0.0f;
    return z;
}

// A-fragment (16x32 f16) from an f16 row-major row (stride-1 in K).
// lane<16:  elems 0..15 -> K k0+0..15
// lane>=16: elems 0..15 -> K k0+16..31
__device__ __forceinline__ v16h load_afrag_f16(const _Float16* row, int k0, int laneHi) {
    return *(const v16h*)(row + k0 + laneHi * 16);
}

// ---------------------------------------------------------------------------
// Weight convert + transpose: wt[n*K + k] = (f16) w[k*N + n]
// ---------------------------------------------------------------------------
__global__ void convert_w_kernel(const float* __restrict__ w, _Float16* __restrict__ wt,
                                 int K, int N) {
    int idx = blockIdx.x * 256 + threadIdx.x;
    if (idx < K * N) {
        int n = idx / K;
        int k = idx - n * K;
        wt[idx] = (_Float16)w[(size_t)k * N + n];
    }
}

// x fp32 -> f16, straight copy (4 elems/thread, vectorized)
__global__ void convert_x_kernel(const float* __restrict__ x, _Float16* __restrict__ xh,
                                 int n4) {
    int idx = blockIdx.x * 256 + threadIdx.x;
    if (idx < n4) {
        v4f v = *(const v4f*)(x + (size_t)idx * 4);
        v4h o;
#pragma unroll
        for (int i = 0; i < 4; ++i) o[i] = (_Float16)v[i];
        *(v4h*)(xh + (size_t)idx * 4) = o;
    }
}

// ---------------------------------------------------------------------------
// QKV projection: [8192 x 1024](f16) x [1024 x 3072] + bias -> Q/K/V f16 [B,H,S,Hd]
// Block tile 64x128, 4 waves of 32x64 each. Fully register-resident.
// ---------------------------------------------------------------------------
__global__ __launch_bounds__(128) void qkv_gemm_kernel(
    const _Float16* __restrict__ xh, const _Float16* __restrict__ wt,
    const float* __restrict__ bias,
    _Float16* __restrict__ qb, _Float16* __restrict__ kb, _Float16* __restrict__ vb)
{
    const int lane   = threadIdx.x & 31;
    const int wave   = threadIdx.x >> 5;       // 0..3
    const int laneHi = lane >> 4;
    const int ln     = lane & 15;

    const int nBlk    = QKV_N / 128;           // 24
    const int bm      = blockIdx.x / nBlk;
    const int bn      = blockIdx.x - bm * nBlk;
    const int rowBase = bm * 64 + (wave >> 1) * 32;
    const int colBase = bn * 128 + (wave & 1) * 64;

    v8f acc[2][4];
#pragma unroll
    for (int r = 0; r < 2; ++r)
#pragma unroll
        for (int g = 0; g < 4; ++g) acc[r][g] = zero8();

    const _Float16* ar0 = xh + (size_t)(rowBase + ln) * DIM;
    const _Float16* ar1 = xh + (size_t)(rowBase + 16 + ln) * DIM;

    for (int k0 = 0; k0 < DIM; k0 += 32) {
        v16h b[4];
#pragma unroll
        for (int g = 0; g < 4; ++g) {
            const _Float16* wrow = wt + (size_t)(colBase + g * 16 + ln) * DIM;
            __builtin_prefetch(wrow + k0 + 64, 0, 3);
            b[g] = *(const v16h*)(wrow + k0 + laneHi * 16);
        }
        v16h a0 = load_afrag_f16(ar0, k0, laneHi);
        v16h a1 = load_afrag_f16(ar1, k0, laneHi);
#pragma unroll
        for (int g = 0; g < 4; ++g) {
            acc[0][g] = wmma_f16(a0, b[g], acc[0][g]);
            acc[1][g] = wmma_f16(a1, b[g], acc[1][g]);
        }
    }

    // Epilogue: bias add, convert f16, scatter into [B,H,S,Hd]
#pragma unroll
    for (int r = 0; r < 2; ++r) {
#pragma unroll
        for (int g = 0; g < 4; ++g) {
            const int n      = colBase + g * 16 + ln;
            const int region = n >> 10;                 // 0=Q 1=K 2=V
            const int jj     = n & 1023;
            const int h      = jj >> 6;
            const int hd     = jj & 63;
            _Float16* dst;
            if      (region == 0) dst = qb;
            else if (region == 1) dst = kb;
            else                  dst = vb;
            const float bv = bias[n];
#pragma unroll
            for (int i = 0; i < 8; ++i) {
                const int m = rowBase + r * 16 + i + laneHi * 8;
                const int b_ = m >> 11;                 // / SEQ
                const int s  = m & (SEQ - 1);
                dst[(((size_t)b_ * NHEADS + h) * SEQ + s) * HDIM + hd] =
                    (_Float16)(acc[r][g][i] + bv);
            }
        }
    }
}

// ---------------------------------------------------------------------------
// Causal flash attention. Block = one (b,h) x 64 query rows; 4 waves x 16 rows.
// K tile staged once per block into LDS (async load-to-LDS path when available),
// V tile staged transposed, P restaged per-wave for the PV WMMAs.
// ---------------------------------------------------------------------------
__global__ __launch_bounds__(128) void attn_kernel(
    const _Float16* __restrict__ qb, const _Float16* __restrict__ kb,
    const _Float16* __restrict__ vb, _Float16* __restrict__ ob)
{
    const int lane   = threadIdx.x & 31;
    const int wave   = threadIdx.x >> 5;
    const int laneHi = lane >> 4;
    const int ln     = lane & 15;

    const int qt = blockIdx.x & 31;          // S/64 = 32 query tiles
    const int bh = blockIdx.x >> 5;          // b*H + h
    const size_t base = (size_t)bh * SEQ * HDIM;
    const int qbase = qt * 64;

    __shared__ __align__(64) _Float16 Klds[32 * HDIM];   // [tok][hd] as-is
    __shared__ __align__(64) _Float16 Vlds[HDIM * 32];   // [hd][tok] transposed
    __shared__ __align__(64) _Float16 Plds[4][16 * 32];  // per-wave P staging

    // Q fragments for this wave's 16 rows (Hd=64 -> two K slices)
    const _Float16* qrow = qb + base + (size_t)(qbase + wave * 16 + ln) * HDIM;
    const v16h qa0 = load_afrag_f16(qrow, 0, laneHi);
    const v16h qa1 = load_afrag_f16(qrow, 32, laneHi);

    v8f oacc[4];
#pragma unroll
    for (int g = 0; g < 4; ++g) oacc[g] = zero8();
    float mrun[8], lrun[8];
#pragma unroll
    for (int i = 0; i < 8; ++i) { mrun[i] = -1e30f; lrun[i] = 0.0f; }

    const float scale = 0.125f;              // 1/sqrt(64)
    const int ktiles = (qbase + 64) / 32;    // keys up to block's last row

    for (int kt = 0; kt < ktiles; ++kt) {
        const int kstart = kt * 32;

        __syncthreads();   // previous iteration's LDS reads complete

        // ---- Stage K tile (32x64 f16 = 4KB) into LDS, layout unchanged ----
        {
            const _Float16* gk = kb + base + (size_t)kstart * HDIM;
#if USE_ASYNC_LDS
#pragma unroll
            for (int c = 0; c < 2; ++c) {
                const int idx16 = threadIdx.x + 128 * c;   // 16-byte chunk id
                __builtin_amdgcn_global_load_async_to_lds_b128(
                    (AS1 v4i*)(gk + idx16 * 8),
                    (AS3 v4i*)(&Klds[idx16 * 8]), 0, 0);
            }
#else
#pragma unroll
            for (int c = 0; c < 2; ++c) {
                const int idx16 = threadIdx.x + 128 * c;
                *(v8h*)(&Klds[idx16 * 8]) = *(const v8h*)(gk + idx16 * 8);
            }
#endif
        }

        // ---- Stage V tile transposed into LDS: Vlds[hd][tok] ----
#pragma unroll
        for (int e = 0; e < 16; ++e) {
            const int id  = threadIdx.x + 128 * e;     // 0..2047
            const int tok = id >> 6;
            const int hd  = id & 63;
            Vlds[hd * 32 + tok] = vb[base + (size_t)(kstart + tok) * HDIM + hd];
        }

#if USE_ASYNC_LDS
        __builtin_amdgcn_s_wait_asynccnt(0);
#endif
        __syncthreads();

        // ---- Scores: 16x32 per wave, 4 WMMAs; B-frags from shared K tile ----
        v8f sc[2] = { zero8(), zero8() };
#pragma unroll
        for (int g = 0; g < 2; ++g) {
            const _Float16* krow = &Klds[(g * 16 + ln) * HDIM];
            v16h b0 = *(const v16h*)(krow + laneHi * 16);        // hd 0..31
            v16h b1 = *(const v16h*)(krow + 32 + laneHi * 16);   // hd 32..63
            sc[g] = wmma_f16(qa0, b0, sc[g]);
            sc[g] = wmma_f16(qa1, b1, sc[g]);
        }

        // ---- Online softmax (rows split across lane halves per C layout) ----
        float alpha[8];
#pragma unroll
        for (int i = 0; i < 8; ++i) {
            const int q = qbase + wave * 16 + i + laneHi * 8;
            float mx = -1e30f;
#pragma unroll
            for (int g = 0; g < 2; ++g) {
                const int col = kstart + g * 16 + ln;
                float s = sc[g][i] * scale;
                if (col > q) s = -1e30f;     // causal mask
                sc[g][i] = s;
                mx = fmaxf(mx, s);
            }
#pragma unroll
            for (int off = 1; off < 16; off <<= 1)
                mx = fmaxf(mx, __shfl_xor(mx, off, 32));
            const float mnew = fmaxf(mrun[i], mx);
            alpha[i] = __expf(mrun[i] - mnew);
            mrun[i] = mnew;
            float ps = 0.0f;
#pragma unroll
            for (int g = 0; g < 2; ++g) {
                float p = __expf(sc[g][i] - mnew);
                sc[g][i] = p;
                ps += p;
            }
#pragma unroll
            for (int off = 1; off < 16; off <<= 1)
                ps += __shfl_xor(ps, off, 32);
            lrun[i] = lrun[i] * alpha[i] + ps;
        }
#pragma unroll
        for (int g = 0; g < 4; ++g)
#pragma unroll
            for (int i = 0; i < 8; ++i) oacc[g][i] *= alpha[i];

        // ---- Restage P (C layout) into A-fragment layout via per-wave LDS ----
#pragma unroll
        for (int g = 0; g < 2; ++g)
#pragma unroll
            for (int i = 0; i < 8; ++i) {
                const int m = i + laneHi * 8;
                Plds[wave][m * 32 + g * 16 + ln] = (_Float16)sc[g][i];
            }
        __syncthreads();

        const v16h pa = load_afrag_f16(&Plds[wave][ln * 32], 0, laneHi);
#pragma unroll
        for (int g = 0; g < 4; ++g) {
            v16h bv = *(const v16h*)(&Vlds[(g * 16 + ln) * 32 + laneHi * 16]);
            oacc[g] = wmma_f16(pa, bv, oacc[g]);
        }
    }

    // Normalize and store to attn buffer [B,S,D] (D = h*64+hd), f16
    const int b_ = bh >> 4;
    const int h  = bh & 15;
#pragma unroll
    for (int i = 0; i < 8; ++i) {
        const float inv = 1.0f / lrun[i];
        const int q = qbase + wave * 16 + i + laneHi * 8;
#pragma unroll
        for (int g = 0; g < 4; ++g) {
            const int hd = g * 16 + ln;
            ob[((size_t)b_ * SEQ + q) * DIM + h * HDIM + hd] =
                (_Float16)(oacc[g][i] * inv);
        }
    }
}

// ---------------------------------------------------------------------------
// Output projection: [8192 x 1024](f16) x [1024 x 1024] + bias -> fp32 out
// ---------------------------------------------------------------------------
__global__ __launch_bounds__(128) void out_gemm_kernel(
    const _Float16* __restrict__ a, const _Float16* __restrict__ wt,
    const float* __restrict__ bias, float* __restrict__ out)
{
    const int lane   = threadIdx.x & 31;
    const int wave   = threadIdx.x >> 5;
    const int laneHi = lane >> 4;
    const int ln     = lane & 15;

    const int nBlk    = DIM / 128;             // 8
    const int bm      = blockIdx.x / nBlk;
    const int bn      = blockIdx.x - bm * nBlk;
    const int rowBase = bm * 64 + (wave >> 1) * 32;
    const int colBase = bn * 128 + (wave & 1) * 64;

    v8f acc[2][4];
#pragma unroll
    for (int r = 0; r < 2; ++r)
#pragma unroll
        for (int g = 0; g < 4; ++g) acc[r][g] = zero8();

    const _Float16* ar0 = a + (size_t)(rowBase + ln) * DIM;
    const _Float16* ar1 = a + (size_t)(rowBase + 16 + ln) * DIM;

    for (int k0 = 0; k0 < DIM; k0 += 32) {
        v16h b[4];
#pragma unroll
        for (int g = 0; g < 4; ++g) {
            const _Float16* wrow = wt + (size_t)(colBase + g * 16 + ln) * DIM;
            __builtin_prefetch(wrow + k0 + 64, 0, 3);
            b[g] = *(const v16h*)(wrow + k0 + laneHi * 16);
        }
        v16h a0 = load_afrag_f16(ar0, k0, laneHi);
        v16h a1 = load_afrag_f16(ar1, k0, laneHi);
#pragma unroll
        for (int g = 0; g < 4; ++g) {
            acc[0][g] = wmma_f16(a0, b[g], acc[0][g]);
            acc[1][g] = wmma_f16(a1, b[g], acc[1][g]);
        }
    }

#pragma unroll
    for (int r = 0; r < 2; ++r)
#pragma unroll
        for (int g = 0; g < 4; ++g) {
            const int n = colBase + g * 16 + ln;
            const float bv = bias[n];
#pragma unroll
            for (int i = 0; i < 8; ++i) {
                const int m = rowBase + r * 16 + i + laneHi * 8;
                out[(size_t)m * DIM + n] = acc[r][g][i] + bv;
            }
        }
}

// ---------------------------------------------------------------------------
// Launch
// ---------------------------------------------------------------------------
extern "C" void kernel_launch(void* const* d_in, const int* in_sizes, int n_in,
                              void* d_out, int out_size, void* d_ws, size_t ws_size,
                              hipStream_t stream) {
    const float* x     = (const float*)d_in[0];
    const float* w_qkv = (const float*)d_in[1];
    const float* b_qkv = (const float*)d_in[2];
    const float* w_out = (const float*)d_in[3];
    const float* b_out = (const float*)d_in[4];
    float* out = (float*)d_out;

    char* ws = (char*)d_ws;
    size_t off = 0;
    _Float16* wqkv_t = (_Float16*)(ws + off); off += (size_t)QKV_N * DIM * 2;   // 6 MB
    _Float16* wout_t = (_Float16*)(ws + off); off += (size_t)DIM * DIM * 2;     // 2 MB
    _Float16* xh     = (_Float16*)(ws + off); off += (size_t)TOKENS * DIM * 2;  // 16 MB
    _Float16* qb     = (_Float16*)(ws + off); off += (size_t)TOKENS * DIM * 2;  // 16 MB
    _Float16* kb     = (_Float16*)(ws + off); off += (size_t)TOKENS * DIM * 2;  // 16 MB
    _Float16* vb     = (_Float16*)(ws + off); off += (size_t)TOKENS * DIM * 2;  // 16 MB
    _Float16* attnb  = (_Float16*)(ws + off); off += (size_t)TOKENS * DIM * 2;  // 16 MB

    // 1) convert weights (transposed, f16 [N][K]) and activations (f16)
    convert_w_kernel<<<(QKV_N * DIM + 255) / 256, 256, 0, stream>>>(w_qkv, wqkv_t, DIM, QKV_N);
    convert_w_kernel<<<(DIM * DIM + 255) / 256, 256, 0, stream>>>(w_out, wout_t, DIM, DIM);
    convert_x_kernel<<<(TOKENS * DIM / 4 + 255) / 256, 256, 0, stream>>>(x, xh, TOKENS * DIM / 4);

    // 2) QKV projection
    qkv_gemm_kernel<<<(TOKENS / 64) * (QKV_N / 128), 128, 0, stream>>>(
        xh, wqkv_t, b_qkv, qb, kb, vb);

    // 3) causal flash attention
    attn_kernel<<<BATCH * NHEADS * (SEQ / 64), 128, 0, stream>>>(qb, kb, vb, attnb);

    // 4) output projection
    out_gemm_kernel<<<(TOKENS / 64) * (DIM / 128), 128, 0, stream>>>(
        attnb, wout_t, b_out, out);
}
